// POS_predictor_43602507989853
// MI455X (gfx1250) — matrix operand order repaired and checked by reading, not compiled, not run on hardware
//
#include <hip/hip_runtime.h>
#include <hip/hip_bf16.h>
#include <cstdint>

// Problem dims
constexpr int NV = 50257;   // vocab
constexpr int NE = 1024;    // embedding
constexpr int NH = 1024;    // hidden
constexpr int NG = 3072;    // 3*NH gates
constexpr int NC = 20;      // classes
constexpr int NB = 64;      // batch
constexpr int NT = 512;     // time steps

typedef __bf16 bf16_t;
typedef __attribute__((ext_vector_type(16))) __bf16 v16bf;
typedef __attribute__((ext_vector_type(8)))  float  v8f;
typedef __attribute__((ext_vector_type(4)))  unsigned int v4u;
typedef __attribute__((ext_vector_type(8)))  int   v8i;
typedef __attribute__((ext_vector_type(4)))  int   v4i;

// Padded LDS row for staged h: 1024 bf16 = 2048B, + 2x16B TDM pad = 2080B.
constexpr int SH_ROW_BYTES = 2080;

// ---------------------------------------------------------------------------
// WMMA fragment loaders (wave32, 16x16x32 bf16).
// A (16x32, MxK): lane<16 -> row=lane, K=[k0..k0+15];
//                 lane>=16 -> row=lane-16, K shifted by +16.
// B (32x16, KxN): lane -> column; lanes<16 K=[k0..k0+15], lanes>=16 +16.
// ---------------------------------------------------------------------------
__device__ __forceinline__ v16bf load_frag_a(const bf16_t* rowbase, int k0, int lane) {
  const int h = lane >> 4;
  const bf16_t* p = rowbase + k0 + h * 16;
  union { v16bf v; uint4 q[2]; } u;
  u.q[0] = *(const uint4*)(p);
  u.q[1] = *(const uint4*)(p + 8);
  return u.v;
}
// A fragment from TDM-staged LDS copy of h (2080B padded rows, pad after
// each 256 dwords -> +16B byte offset for the upper K half).
__device__ __forceinline__ v16bf load_frag_a_lds(const char* base, int row, int k0, int lane) {
  const int h = lane >> 4;
  const char* p = base + row * SH_ROW_BYTES + ((k0 >= 512) ? 16 : 0) + (k0 + h * 16) * 2;
  union { v16bf v; uint4 q[2]; } u;
  u.q[0] = *(const uint4*)(p);
  u.q[1] = *(const uint4*)(p + 16);
  return u.v;
}
// B fragment from fragment-major packed weights: fragment (ntile, kchunk) is a
// contiguous 1KB block; lane i reads bytes [32i, 32i+32).
__device__ __forceinline__ v16bf load_frag_b_pk(const bf16_t* pack, int gnt, int kc, int KC, int lane) {
  const bf16_t* p = pack + ((size_t)(gnt * KC + kc) * 512) + lane * 16;
  union { v16bf v; uint4 q[2]; } u;
  u.q[0] = *(const uint4*)(p);
  u.q[1] = *(const uint4*)(p + 8);
  return u.v;
}
#define WMMA_BF16(a, b, c) \
  __builtin_amdgcn_wmma_f32_16x16x32_bf16(false, (a), false, (b), (short)0, (c), false, false)

// ---------------------------------------------------------------------------
// Prep kernels
// ---------------------------------------------------------------------------
__global__ void cvt_f32_to_bf16(const float* __restrict__ src, bf16_t* __restrict__ dst, int n) {
  int i = blockIdx.x * blockDim.x + threadIdx.x;
  int stride = gridDim.x * blockDim.x;
  for (; i < n; i += stride) dst[i] = (bf16_t)src[i];
}

// Pack row-major f32 [N,K] weights into WMMA-B fragment-major bf16 layout.
// One thread per (ntile, kchunk, lane) writes its 16 contiguous elements.
// Rows >= Nreal are zero (classifier padding).
__global__ void pack_b_frag(const float* __restrict__ src, bf16_t* __restrict__ dst,
                            int N, int K, int Nreal) {
  int i = blockIdx.x * blockDim.x + threadIdx.x;
  int total = (N / 16) * (K / 32) * 32;
  if (i >= total) return;
  const int lane = i & 31;
  const int kc = (i >> 5) % (K / 32);
  const int nt = (i >> 5) / (K / 32);
  const int row = nt * 16 + (lane & 15);
  const int kbase = kc * 32 + (lane >> 4) * 16;
  bf16_t* d = dst + (size_t)i * 16;
  if (row < Nreal) {
    const float* s = src + (size_t)row * K + kbase;
#pragma unroll
    for (int e = 0; e < 16; ++e) d[e] = (bf16_t)s[e];
  } else {
#pragma unroll
    for (int e = 0; e < 16; ++e) d[e] = (bf16_t)0.0f;
  }
}

__global__ void zero_bf16_k(bf16_t* p, int n) {
  int i = blockIdx.x * blockDim.x + threadIdx.x;
  if (i < n) p[i] = (bf16_t)0.0f;
}
__global__ void zero_f32_k(float* p, int n) {
  int i = blockIdx.x * blockDim.x + threadIdx.x;
  if (i < n) p[i] = 0.0f;
}

// ---------------------------------------------------------------------------
// Phase A: gi_all[bt, g] = emb[x[bt]] . w_ih[g] + b_ih[g]
// M = 32768, N = 3072, K = 1024. Wave tile = 32x64 (2 M x 4 N subtiles).
// grid = (NG/512, M/32) = (6, 1024), block = 8 waves.
// ---------------------------------------------------------------------------
__global__ void __launch_bounds__(256) gi_gemm(
    const int* __restrict__ x, const bf16_t* __restrict__ embb,
    const bf16_t* __restrict__ wih_pk, const float* __restrict__ b_ih,
    float* __restrict__ gi) {
  const int lane = threadIdx.x & 31;
  const int wave = threadIdx.x >> 5;
  const int halfl = lane >> 4;
  const int idx = lane & 15;
  const int mbase = blockIdx.y * 32;
  const int ncol0 = blockIdx.x * 512 + wave * 64;
  const int nt0 = ncol0 >> 4;

  const bf16_t* arow0 = embb + (size_t)x[mbase + idx] * NE;
  const bf16_t* arow1 = embb + (size_t)x[mbase + 16 + idx] * NE;

  v8f acc[8] = {};
#pragma unroll 2
  for (int kc = 0; kc < NE / 32; ++kc) {
    const int k0 = kc * 32;
    v16bf a0 = load_frag_a(arow0, k0, lane);
    v16bf a1 = load_frag_a(arow1, k0, lane);
#pragma unroll
    for (int nt = 0; nt < 4; ++nt) {
      v16bf b = load_frag_b_pk(wih_pk, nt0 + nt, kc, NE / 32, lane);
      acc[nt * 2]     = WMMA_BF16(a0, b, acc[nt * 2]);
      acc[nt * 2 + 1] = WMMA_BF16(a1, b, acc[nt * 2 + 1]);
    }
  }
#pragma unroll
  for (int nt = 0; nt < 4; ++nt) {
    const int col = ncol0 + nt * 16 + idx;
    const float bias = b_ih[col];
#pragma unroll
    for (int s = 0; s < 2; ++s) {
#pragma unroll
      for (int r = 0; r < 8; ++r) {
        const int mm = mbase + s * 16 + r + 8 * halfl;
        gi[(size_t)mm * NG + col] = acc[nt * 2 + s][r] + bias;
      }
    }
  }
}

// ---------------------------------------------------------------------------
// Phase B: one GRU timestep. 16 blocks; block owns hidden slice j0..j0+63
// (192 gate columns). h_t is staged into LDS by the Tensor Data Mover
// (padded rows to spread LDS banks), gh goes through LDS (overlaid on h
// after the GEMM), then fused gate math produces h_{t+1}.
// hsbuf: [NT+1][NB][NH] bf16 (slot t = h_t; slot 0 zeroed).
// Dynamic LDS: 64 * 2080 = 133120 bytes.
// ---------------------------------------------------------------------------
__global__ void __launch_bounds__(256) gru_step(
    int t, const float* __restrict__ gi, const bf16_t* __restrict__ whh_pk,
    const float* __restrict__ b_hh, bf16_t* hsbuf,
    const float* __restrict__ hprev, float* __restrict__ hnext) {
  extern __shared__ __align__(16) char smem[];
  const int lane = threadIdx.x & 31;
  const int wave = threadIdx.x >> 5;
  const int halfl = lane >> 4;
  const int idx = lane & 15;
  const int j0 = blockIdx.x * 64;

  // --- TDM: stage h_t [64 x 1024] bf16 into LDS with padded rows ---------
  // Group1 dword0: data_size=2B (1<<16), pad_enable (1<<20),
  //                pad_interval=256 dwords (7<<22), pad_amount=4 dwords (3<<25)
  if (threadIdx.x < 32) {
    const uint64_t gaddr = (uint64_t)(uintptr_t)(hsbuf + (size_t)t * (NB * NH));
    const unsigned int lds_off = (unsigned int)(uintptr_t)(&smem[0]);
    v4u g0 = { 1u,                                  // count=1 user descriptor
               lds_off,                             // lds_addr
               (unsigned int)gaddr,                 // global_addr[31:0]
               (unsigned int)((gaddr >> 32) & 0x1FFFFFFu) | (2u << 30) };  // [56:32] | type=2
    v8i g1 = { (int)((1u << 16) | (1u << 20) | (7u << 22) | (3u << 25)),
               (int)(1024u << 16),                  // tensor_dim0 = 1024 (lo16)
               (int)(64u << 16),                    // tensor_dim1 = 64 (lo16)
               (int)(1024u << 16),                  // tile_dim0 = 1024
               64,                                  // tile_dim1 = 64
               1024,                                // tensor_dim0_stride
               0, 0 };
    v4i gz4 = { 0, 0, 0, 0 };
    v8i gz8 = { 0, 0, 0, 0, 0, 0, 0, 0 };
    __builtin_amdgcn_tensor_load_to_lds(g0, g1, gz4, gz4, gz8, 0);
    __builtin_amdgcn_s_wait_tensorcnt(0);
  }
  __syncthreads();

  // --- GEMM: gh = h_t @ w_hh^T for this block's 192 gate columns ---------
  const int mt = wave & 3;              // 16 batch rows
  const int ntbase = (wave >> 2) * 6;   // 6 of the 12 N tiles
  int gnt[6];
#pragma unroll
  for (int q = 0; q < 6; ++q) {
    const int nt = ntbase + q;
    gnt[q] = (nt >> 2) * 64 + (j0 >> 4) + (nt & 3);  // global w_hh ntile
  }
  const int arow = mt * 16 + idx;

  v8f acc[6] = {};
#pragma unroll 2
  for (int kc = 0; kc < NH / 32; ++kc) {
    const int k0 = kc * 32;
    v16bf a = load_frag_a_lds(smem, arow, k0, lane);
#pragma unroll
    for (int q = 0; q < 6; ++q) {
      v16bf b = load_frag_b_pk(whh_pk, gnt[q], kc, NH / 32, lane);
      acc[q] = WMMA_BF16(a, b, acc[q]);
    }
  }
  __syncthreads();                      // h staging dead; overlay gh

  float* gh = (float*)smem;             // [64][192]
#pragma unroll
  for (int q = 0; q < 6; ++q) {
    const int col = (ntbase + q) * 16 + idx;
#pragma unroll
    for (int r = 0; r < 8; ++r) {
      const int mm = mt * 16 + r + 8 * halfl;
      gh[mm * 192 + col] = acc[q][r];
    }
  }
  __syncthreads();

  // --- gate math: 64 batch x 64 hidden, 16 per thread --------------------
  const int bidx = threadIdx.x >> 2;
  const int jbase = (threadIdx.x & 3) * 16;
  const float* gib = gi + ((size_t)bidx * NT + t) * NG + j0;
  bf16_t* hs_out = hsbuf + (size_t)(t + 1) * (NB * NH);
#pragma unroll
  for (int q = 0; q < 16; ++q) {
    const int j = jbase + q;
    const int gj = j0 + j;
    const float ghr = gh[bidx * 192 + j]       + b_hh[gj];
    const float ghz = gh[bidx * 192 + 64 + j]  + b_hh[NH + gj];
    const float ghn = gh[bidx * 192 + 128 + j] + b_hh[2 * NH + gj];
    const float rg = 1.0f / (1.0f + __expf(-(gib[j] + ghr)));
    const float zg = 1.0f / (1.0f + __expf(-(gib[NH + j] + ghz)));
    const float ng = tanhf(gib[2 * NH + j] + rg * ghn);
    const float hp = hprev[bidx * NH + gj];
    const float hn = (1.0f - zg) * ng + zg * hp;
    hnext[bidx * NH + gj] = hn;
    hs_out[(size_t)bidx * NH + gj] = (bf16_t)hn;
  }
}

// ---------------------------------------------------------------------------
// Phase C: pred[bt, c] = hs[t+1][b] . cls_w[c] + cls_b[c]
// M = 32768, N = 32 (20 real + 12 zero rows), K = 1024. grid = 256.
// ---------------------------------------------------------------------------
__global__ void __launch_bounds__(256) cls_gemm(
    const bf16_t* __restrict__ hsbuf, const bf16_t* __restrict__ cls_pk,
    const float* __restrict__ cls_b, float* __restrict__ out) {
  const int lane = threadIdx.x & 31;
  const int wave = threadIdx.x >> 5;
  const int halfl = lane >> 4;
  const int idx = lane & 15;
  const int mtile = blockIdx.x * 8 + wave;
  const int m = mtile * 16 + idx;
  const int bb = m / NT, tt = m % NT;
  const bf16_t* arow = hsbuf + ((size_t)(tt + 1) * NB + bb) * NH;

  v8f acc[2] = {};
#pragma unroll 4
  for (int kc = 0; kc < NH / 32; ++kc) {
    const int k0 = kc * 32;
    v16bf a = load_frag_a(arow, k0, lane);
#pragma unroll
    for (int nt = 0; nt < 2; ++nt) {
      v16bf b = load_frag_b_pk(cls_pk, nt, kc, NH / 32, lane);
      acc[nt] = WMMA_BF16(a, b, acc[nt]);
    }
  }
#pragma unroll
  for (int nt = 0; nt < 2; ++nt) {
    const int col = nt * 16 + idx;
    if (col < NC) {
      const float bias = cls_b[col];
#pragma unroll
      for (int r = 0; r < 8; ++r) {
        const int mm = mtile * 16 + r + 8 * halfl;
        out[(size_t)mm * NC + col] = acc[nt][r] + bias;
      }
    }
  }
}

// ---------------------------------------------------------------------------
extern "C" void kernel_launch(void* const* d_in, const int* in_sizes, int n_in,
                              void* d_out, int out_size, void* d_ws, size_t ws_size,
                              hipStream_t stream) {
  const int*   x     = (const int*)  d_in[0];
  const float* emb   = (const float*)d_in[1];
  const float* w_ih  = (const float*)d_in[2];
  const float* w_hh  = (const float*)d_in[3];
  const float* b_ih  = (const float*)d_in[4];
  const float* b_hh  = (const float*)d_in[5];
  const float* cls_w = (const float*)d_in[6];
  const float* cls_b = (const float*)d_in[7];
  float* out = (float*)d_out;

  char* ws = (char*)d_ws;
  size_t off = 0;
  auto walloc = [&](size_t bytes) -> void* {
    void* p = ws + off;
    off = (off + bytes + 255) & ~(size_t)255;
    return p;
  };
  bf16_t* embb   = (bf16_t*)walloc((size_t)NV * NE * sizeof(bf16_t));
  bf16_t* wih_pk = (bf16_t*)walloc((size_t)NG * NE * sizeof(bf16_t));
  bf16_t* whh_pk = (bf16_t*)walloc((size_t)NG * NH * sizeof(bf16_t));
  bf16_t* cls_pk = (bf16_t*)walloc((size_t)32 * NH * sizeof(bf16_t));
  float*  gi     = (float*) walloc((size_t)NB * NT * NG * sizeof(float));
  bf16_t* hsbuf  = (bf16_t*)walloc((size_t)(NT + 1) * NB * NH * sizeof(bf16_t));
  float*  hf32   = (float*) walloc((size_t)2 * NB * NH * sizeof(float));
  (void)ws_size; (void)in_sizes; (void)n_in; (void)out_size;

  // --- prep: emb -> bf16, weights -> fragment-major bf16, zero h0 ---------
  cvt_f32_to_bf16<<<4096, 256, 0, stream>>>(emb, embb, NV * NE);
  {
    int tot = (NG / 16) * (NE / 32) * 32;
    pack_b_frag<<<(tot + 255) / 256, 256, 0, stream>>>(w_ih, wih_pk, NG, NE, NG);
  }
  {
    int tot = (NG / 16) * (NH / 32) * 32;
    pack_b_frag<<<(tot + 255) / 256, 256, 0, stream>>>(w_hh, whh_pk, NG, NH, NG);
  }
  {
    int tot = (32 / 16) * (NH / 32) * 32;
    pack_b_frag<<<(tot + 255) / 256, 256, 0, stream>>>(cls_w, cls_pk, 32, NH, NC);
  }
  zero_bf16_k<<<(NB * NH + 255) / 256, 256, 0, stream>>>(hsbuf, NB * NH);
  zero_f32_k<<<(NB * NH + 255) / 256, 256, 0, stream>>>(hf32, NB * NH);

  // --- phase A: big input-projection GEMM ---------------------------------
  gi_gemm<<<dim3(6, (NB * NT) / 32), 256, 0, stream>>>(x, embb, wih_pk, b_ih, gi);

  // --- phase B: sequential recurrence, one launch per timestep ------------
  const size_t shbytes = (size_t)NB * SH_ROW_BYTES;  // 133120
  for (int t = 0; t < NT; ++t) {
    float* hp = hf32 + (size_t)(t & 1) * NB * NH;
    float* hn = hf32 + (size_t)((t + 1) & 1) * NB * NH;
    gru_step<<<16, 256, shbytes, stream>>>(t, gi, whh_pk, b_hh, hsbuf, hp, hn);
  }

  // --- phase C: classifier GEMM -------------------------------------------
  cls_gemm<<<(NB * NT) / 16 / 8, 256, 0, stream>>>(hsbuf, cls_pk, cls_b, out);
}
